// Attention_30657476559131
// MI455X (gfx1250) — matrix operand order repaired and checked by reading, not compiled, not run on hardware
//
#include <hip/hip_runtime.h>
#include <stdint.h>

// ---------------------------------------------------------------------------
// ViT window attention w/ decomposed rel-pos bias + policy-masked softmax.
// MI455X (gfx1250): bf16 WMMA (v_wmma_f32_16x16x32_bf16) for all GEMMs,
// fused flash-attention (the 96x1024x1024 score tensor never touches HBM:
// ~800 MB of traffic saved vs materializing at 23.3 TB/s), V stored
// transposed so P@V B-operands are contiguous, P transposed through LDS.
// Softmax bookkeeping optimized for wave32:
//   * row-sums via an all-ones WMMA B tile (no cross-lane sum shuffles)
//   * per-tile (not per-row) running max: any per-row-uniform shift keeps
//     softmax exact; 5 shuffles + 1 exp per 32-key step
//   * diagonal policy fixup only in the single 32-key step that contains it
// Workspace requirement: ~93 MB.
// ---------------------------------------------------------------------------

typedef __attribute__((ext_vector_type(16))) __bf16 v16bf;
typedef __attribute__((ext_vector_type(8)))  float  v8f;

#define DIM   768
#define NHEAD 12
#define HD    64
#define NTOK  1024
#define BATCH 8
#define BN    (BATCH * NTOK)    // 8192 rows
#define HEADS (BATCH * NHEAD)   // 96 attention problems
#define QKVN  (3 * DIM)         // 2304

struct BfPack { uint4 lo, hi; };

static __device__ inline uint16_t f2bf(float f) {
  uint32_t x = __builtin_bit_cast(uint32_t, f);
  x += 0x7FFFu + ((x >> 16) & 1u);          // round-to-nearest-even
  return (uint16_t)(x >> 16);
}
static __device__ inline float bf2f(uint16_t h) {
  return __builtin_bit_cast(float, (uint32_t)h << 16);
}

// WMMA A operand (16x32 bf16, M x K). ISA 7.12.2:
//   lanes 0-15 : row = lane,    K = {k0..k0+7, k0+16..k0+23}
//   lanes 16-31: row = lane-16, K = {k0+8..k0+15, k0+24..k0+31}
static __device__ inline v16bf load_a(const uint16_t* __restrict__ base, int stride,
                                      int row, int k0, int hi) {
  const uint16_t* p = base + (size_t)row * stride + k0 + hi * 8;
  BfPack u; u.lo = *(const uint4*)p; u.hi = *(const uint4*)(p + 16);
  return __builtin_bit_cast(v16bf, u);
}

// WMMA B operand (32x16 bf16, K x N), same K order as A:
//   lanes 0-15 : col = lane,    K = {k0..k0+7, k0+16..k0+23}
//   lanes 16-31: col = lane-16, K = {k0+8..k0+15, k0+24..k0+31}
// `base` stores columns of B contiguously along K (row-major [col][K]).
static __device__ inline v16bf load_b(const uint16_t* __restrict__ base, int stride,
                                      int col, int k0, int hi) {
  const uint16_t* p = base + (size_t)col * stride + k0 + hi * 8;
  BfPack u; u.lo = *(const uint4*)p; u.hi = *(const uint4*)(p + 16);
  return __builtin_bit_cast(v16bf, u);
}

static __device__ inline v8f wmma_bf16(v16bf a, v16bf b, v8f c) {
  return __builtin_amdgcn_wmma_f32_16x16x32_bf16(false, a, false, b,
                                                 (short)0, c, false, false);
}

// ---------------------------------------------------------------------------
__global__ void cvt_bf16(const float* __restrict__ in, uint16_t* __restrict__ out, int n) {
  int i = blockIdx.x * 256 + threadIdx.x;
  if (i < n) out[i] = f2bf(in[i]);
}

// ---------------------------------------------------------------------------
// Generic bf16 WMMA GEMM: C[m, c] = sum_k A[m, k] * W[c, k]  (+ bias[c])
// 4 waves/block; each wave owns a 16(M) x 64(N) strip; K = 768 in steps of 32.
// EPI 0: QKV epilogue -> scatter q (per head), k (*1/8), vT (transposed).
// EPI 1: proj epilogue -> f32 out[m*768 + c].
template <int EPI>
__global__ __launch_bounds__(128) void gemm_kernel(
    const uint16_t* __restrict__ A, const uint16_t* __restrict__ W,
    const float* __restrict__ bias,
    uint16_t* __restrict__ qout, uint16_t* __restrict__ kout,
    uint16_t* __restrict__ vtout, float* __restrict__ fout) {
  const int wave = threadIdx.x >> 5;
  const int lane = threadIdx.x & 31;
  const int l16  = lane & 15;
  const int hi   = lane >> 4;
  const int m0 = blockIdx.y * 64 + wave * 16;
  const int n0 = blockIdx.x * 64;

  v8f acc[4];
#pragma unroll
  for (int t = 0; t < 4; ++t)
#pragma unroll
    for (int i = 0; i < 8; ++i) acc[t][i] = 0.0f;

  for (int k0 = 0; k0 < DIM; k0 += 32) {
    if (k0 + 64 <= DIM - 32) {   // pull next K slab toward the WGP (global_prefetch_b8)
      __builtin_prefetch(A + (size_t)(m0 + l16) * DIM + k0 + 64);
      __builtin_prefetch(W + (size_t)(n0 + lane) * DIM + k0 + 64);
    }
    v16bf a = load_a(A, DIM, m0 + l16, k0, hi);
#pragma unroll
    for (int t = 0; t < 4; ++t) {
      v16bf b = load_b(W, DIM, n0 + t * 16 + l16, k0, hi);
      acc[t] = wmma_bf16(a, b, acc[t]);
    }
  }

  const int mrow = m0 + hi * 8;   // C layout: vgpr i -> row mrow+i, col = n-tile + l16
#pragma unroll
  for (int t = 0; t < 4; ++t) {
    const int c  = n0 + t * 16 + l16;
    const float bb = bias[c];
    if (EPI == 0) {
      const int sec = c / DIM;          // 0=q 1=k 2=v (64-wide tiles never straddle)
      const int cc  = c - sec * DIM;
      const int h   = cc >> 6;
      const int d   = cc & 63;
#pragma unroll
      for (int i = 0; i < 8; ++i) {
        const int m = mrow + i;
        const int bidx = m >> 10;
        const int n    = m & (NTOK - 1);
        const int head = bidx * NHEAD + h;
        const float v  = acc[t][i] + bb;
        if (sec == 0)       qout[((size_t)head * NTOK + n) * HD + d] = f2bf(v);
        else if (sec == 1)  kout[((size_t)head * NTOK + n) * HD + d] = f2bf(v * 0.125f);
        else                vtout[((size_t)head * HD + d) * NTOK + n] = f2bf(v);
      }
    } else {
#pragma unroll
      for (int i = 0; i < 8; ++i)
        fout[(size_t)(mrow + i) * DIM + c] = acc[t][i] + bb;
    }
  }
}

// ---------------------------------------------------------------------------
// vsum[head][d] = sum_n V[head][n][d]  (V stored transposed: vT[head][d][n])
__global__ void vsum_kernel(const uint16_t* __restrict__ vt, float* __restrict__ vs) {
  int gid = blockIdx.x * 256 + threadIdx.x;
  if (gid >= HEADS * HD) return;
  const uint16_t* p = vt + (size_t)gid * NTOK;
  float s = 0.0f;
  for (int n = 0; n < NTOK; ++n) s += bf2f(p[n]);
  vs[gid] = s;
}

// ---------------------------------------------------------------------------
// rel_h[bh][n][k] = sum_c q[bh][n][c] * rel_pos_h[h - k + 31][c]   (h = n>>5)
// rel_w[bh][n][k] = sum_c q[bh][n][c] * rel_pos_w[w - k + 31][c]   (w = n&31)
__global__ void rel_kernel(const uint16_t* __restrict__ qb,
                           const float* __restrict__ rph, const float* __restrict__ rpw,
                           float* __restrict__ relh, float* __restrict__ relw) {
  int gid = blockIdx.x * 256 + threadIdx.x;
  const int HALF = HEADS * NTOK * 32;           // 3145728
  if (gid >= 2 * HALF) return;
  const int which = gid >= HALF;
  int r = which ? gid - HALF : gid;
  const int k  = r & 31;
  const int n  = (r >> 5) & (NTOK - 1);
  const int bh = r >> 15;
  const int hh = n >> 5, ww = n & 31;
  const int idx = (which ? ww : hh) - k + 31;   // 0..62
  const float* R = (which ? rpw : rph) + (size_t)idx * HD;
  const uint16_t* qrow = qb + ((size_t)bh * NTOK + n) * HD;
  float acc = 0.0f;
#pragma unroll 8
  for (int c = 0; c < HD; ++c) acc += bf2f(qrow[c]) * R[c];
  (which ? relw : relh)[((size_t)bh * NTOK + n) * 32 + k] = acc;
}

// ---------------------------------------------------------------------------
// Fused flash attention: grid (16 q-blocks, 96 heads), 4 waves x 16 queries.
// Per 32-key step: S = Qs @ K^T (4 wmma) + rel bias, tile-max online softmax,
// P via LDS transpose, O += P @ [V | 1] (5 wmma, 5th tile = ones -> row sums).
__global__ __launch_bounds__(128) void attn_kernel(
    const uint16_t* __restrict__ qb, const uint16_t* __restrict__ kb,
    const uint16_t* __restrict__ vt, const float* __restrict__ relh,
    const float* __restrict__ relw, const float* __restrict__ pol,
    const float* __restrict__ vs, uint16_t* __restrict__ ob) {
  __shared__ __align__(16) uint16_t pst[4][16 * 32];   // per-wave P staging

  const int wave = threadIdx.x >> 5;
  const int lane = threadIdx.x & 31;
  const int l16  = lane & 15;
  const int hi   = lane >> 4;
  const int bh = blockIdx.y;
  const int b  = bh / NHEAD;
  const int h  = bh - b * NHEAD;

  const uint16_t* qh = qb + (size_t)bh * NTOK * HD;
  const uint16_t* kh = kb + (size_t)bh * NTOK * HD;
  const uint16_t* vh = vt + (size_t)bh * HD * NTOK;
  const float* ph = pol + (size_t)b * NTOK;
  const float* rh = relh + (size_t)bh * NTOK * 32;
  const float* rw = relw + (size_t)bh * NTOK * 32;

  const int q0   = blockIdx.x * 64 + wave * 16;
  const int rowb = q0 + hi * 8;                 // C-layout row base for this lane
  const int diagblk = q0 & ~31;                 // only 32-key step containing key==q
  uint16_t* myst = pst[wave];

  const v16bf aq0 = load_a(qh, HD, q0 + l16, 0, hi);
  const v16bf aq1 = load_a(qh, HD, q0 + l16, 32, hi);

  v16bf ones;
#pragma unroll
  for (int i = 0; i < 16; ++i) ones[i] = (__bf16)1.0f;

  // rel_w bias is loop-invariant: kw = key & 31 = l16 (+16) for every 32-key step
  float rwb0[8], rwb1[8];
#pragma unroll
  for (int i = 0; i < 8; ++i) {
    const int row = rowb + i;
    rwb0[i] = rw[(size_t)row * 32 + l16];
    rwb1[i] = rw[(size_t)row * 32 + l16 + 16];
  }

  float mtile = -__builtin_inff();              // per-tile running shift
  v8f acc[5];                                   // [0..3] = O, [4] = row sums (ones tile)
#pragma unroll
  for (int t = 0; t < 5; ++t)
#pragma unroll
    for (int i = 0; i < 8; ++i) acc[t][i] = 0.0f;

  for (int kbase = 0; kbase < NTOK; kbase += 32) {
    // ---- S = (Q/sqrt(d)) @ K^T, two 16x16 key tiles -----------------------
    v16bf b00 = load_b(kh, HD, kbase + l16,      0,  hi);
    v16bf b01 = load_b(kh, HD, kbase + l16,      32, hi);
    v16bf b10 = load_b(kh, HD, kbase + 16 + l16, 0,  hi);
    v16bf b11 = load_b(kh, HD, kbase + 16 + l16, 32, hi);
    v8f z;
#pragma unroll
    for (int i = 0; i < 8; ++i) z[i] = 0.0f;
    v8f s0 = wmma_bf16(aq0, b00, z); s0 = wmma_bf16(aq1, b01, s0);
    v8f s1 = wmma_bf16(aq0, b10, z); s1 = wmma_bf16(aq1, b11, s1);

    const int khh  = kbase >> 5;
    const int key0 = kbase + l16, key1 = key0 + 16;
    const float w0 = ph[key0], w1 = ph[key1];

    // bias + per-lane max over this lane's 16 elements
    float s0f[8], s1f[8];
    float t = -__builtin_inff();
#pragma unroll
    for (int i = 0; i < 8; ++i) {
      const float bias_h = rh[(size_t)(rowb + i) * 32 + khh];
      s0f[i] = s0[i] + bias_h + rwb0[i];
      s1f[i] = s1[i] + bias_h + rwb1[i];
      t = fmaxf(t, fmaxf(s0f[i], s1f[i]));
    }
    // full-wave tile max (uniform shift keeps softmax exact; only guards overflow)
#pragma unroll
    for (int m = 16; m >= 1; m >>= 1) t = fmaxf(t, __shfl_xor(t, m));
    const float mn   = fmaxf(mtile, t);
    const float corr = __expf(mtile - mn);
    mtile = mn;

    // e = exp(s - m) * max(policy[key], key==q); diagonal fixup in one step only
    if (kbase == diagblk) {
#pragma unroll
      for (int i = 0; i < 8; ++i) {
        const int row = rowb + i;
        const float e0 = __expf(s0f[i] - mn) * ((key0 == row) ? 1.0f : w0);
        const float e1 = __expf(s1f[i] - mn) * ((key1 == row) ? 1.0f : w1);
        myst[(hi * 8 + i) * 32 + l16]      = f2bf(e0);
        myst[(hi * 8 + i) * 32 + 16 + l16] = f2bf(e1);
      }
    } else {
#pragma unroll
      for (int i = 0; i < 8; ++i) {
        const float e0 = __expf(s0f[i] - mn) * w0;
        const float e1 = __expf(s1f[i] - mn) * w1;
        myst[(hi * 8 + i) * 32 + l16]      = f2bf(e0);
        myst[(hi * 8 + i) * 32 + 16 + l16] = f2bf(e1);
      }
    }

    // rescale O + rowsum accumulators by the (uniform) correction
#pragma unroll
    for (int t5 = 0; t5 < 5; ++t5) acc[t5] *= corr;

    // read P back in A-operand layout (same-wave DS ops are in-order)
    asm volatile("s_wait_dscnt 0" ::: "memory");
    const uint16_t* pp = myst + l16 * 32 + hi * 8;
    BfPack u; u.lo = *(const uint4*)pp; u.hi = *(const uint4*)(pp + 16);
    const v16bf pa = __builtin_bit_cast(v16bf, u);
    asm volatile("" ::: "memory");

    // O += P @ V ; vT[head][d][n] makes B-operand loads contiguous along keys
#pragma unroll
    for (int dt = 0; dt < 4; ++dt) {
      v16bf bv = load_b(vh, NTOK, dt * 16 + l16, kbase, hi);
      acc[dt] = wmma_bf16(pa, bv, acc[dt]);
    }
    acc[4] = wmma_bf16(pa, ones, acc[4]);       // row sums: l = l*corr + sum(P)
  }

  // out = (acc + (EPS/N)*sum_k v) / (l + EPS), scattered to [b][n][h*64+d]
  const float epsn = 1e-6f / (float)NTOK;
#pragma unroll
  for (int i = 0; i < 8; ++i) {
    const float rinv = 1.0f / (acc[4][i] + 1e-6f);
    const int row = rowb + i;
#pragma unroll
    for (int dt = 0; dt < 4; ++dt) {
      const int d = dt * 16 + l16;
      const float val = (acc[dt][i] + epsn * vs[bh * HD + d]) * rinv;
      ob[((size_t)(b * NTOK + row)) * DIM + h * HD + d] = f2bf(val);
    }
  }
}

// ---------------------------------------------------------------------------
extern "C" void kernel_launch(void* const* d_in, const int* in_sizes, int n_in,
                              void* d_out, int out_size, void* d_ws, size_t ws_size,
                              hipStream_t stream) {
  (void)in_sizes; (void)n_in; (void)out_size;
  const float* x     = (const float*)d_in[0];
  const float* pol   = (const float*)d_in[1];
  const float* qkvw  = (const float*)d_in[2];
  const float* qkvb  = (const float*)d_in[3];
  const float* projw = (const float*)d_in[4];
  const float* projb = (const float*)d_in[5];
  const float* rph   = (const float*)d_in[6];
  const float* rpw   = (const float*)d_in[7];

  uint8_t* ws = (uint8_t*)d_ws;                 // needs ~92.9 MB
  uint16_t* xb   = (uint16_t*)(ws + 0);         // x bf16        [8192][768]
  uint16_t* wb   = (uint16_t*)(ws + 12582912);  // qkv_w bf16    [2304][768]
  uint16_t* pb   = (uint16_t*)(ws + 16121856);  // proj_w bf16   [768][768]
  uint16_t* qb   = (uint16_t*)(ws + 17301504);  // q bf16        [96][1024][64]
  uint16_t* kbuf = (uint16_t*)(ws + 29884416);  // k*scale bf16  [96][1024][64]
  uint16_t* vt   = (uint16_t*)(ws + 42467328);  // v^T bf16      [96][64][1024]
  float*    relh = (float*)  (ws + 55050240);   // f32           [96][1024][32]
  float*    relw = (float*)  (ws + 67633152);   // f32           [96][1024][32]
  float*    vs   = (float*)  (ws + 80216064);   // f32           [96][64]
  uint16_t* ob   = (uint16_t*)(ws + 80240640);  // attn out bf16 [8192][768]
  if (ws_size < 92823552) return;

  cvt_bf16<<<(BN * DIM + 255) / 256, 256, 0, stream>>>(x, xb, BN * DIM);
  cvt_bf16<<<(QKVN * DIM + 255) / 256, 256, 0, stream>>>(qkvw, wb, QKVN * DIM);
  cvt_bf16<<<(DIM * DIM + 255) / 256, 256, 0, stream>>>(projw, pb, DIM * DIM);

  gemm_kernel<0><<<dim3(QKVN / 64, BN / 64), 128, 0, stream>>>(
      xb, wb, qkvb, qb, kbuf, vt, nullptr);

  vsum_kernel<<<(HEADS * HD + 255) / 256, 256, 0, stream>>>(vt, vs);
  rel_kernel<<<(2 * HEADS * NTOK * 32 + 255) / 256, 256, 0, stream>>>(
      qb, rph, rpw, relh, relw);

  attn_kernel<<<dim3(NTOK / 64, HEADS), 128, 0, stream>>>(
      qb, kbuf, vt, relh, relw, pol, vs, ob);

  gemm_kernel<1><<<dim3(DIM / 64, BN / 64), 128, 0, stream>>>(
      ob, pb, projb, nullptr, nullptr, nullptr, (float*)d_out);
}
